// Decoder_21423296873105
// MI455X (gfx1250) — matrix operand order repaired and checked by reading, not compiled, not run on hardware
//
#include <hip/hip_runtime.h>

typedef __attribute__((ext_vector_type(16))) _Float16 v16h;
typedef __attribute__((ext_vector_type(8)))  _Float16 v8h;
typedef __attribute__((ext_vector_type(8)))  float    v8f;

#define HDIM 512
#define DDIM 513
#define GDIM 2048
#define WIH_STRIDE 520   // padded f16 row stride for W_ih (mult of 8 -> 16B aligned chunks)
#define FCW_ROWS   528   // 513 valid rows + zero padding to a full 33rd N-tile
#define XH_STRIDE  528   // LDS X tile stride in halves
#define HH_STRIDE  520   // LDS h_new tile stride in halves
#define RT 32            // rows per block (two 16-row WMMA tiles)
#define NT_OUT 33        // ceil(513/16) output N-tiles

// ---------------- device helpers ----------------

__device__ __forceinline__ float sigmoid_(float x) {
    return 1.0f / (1.0f + __expf(-x));
}
__device__ __forceinline__ float tanh_(float x) {
    // robust at +/- inf arguments of exp
    return 1.0f - 2.0f / (__expf(2.0f * x) + 1.0f);
}

// A fragment (16x32, f16). Per ISA: lane m=lane&15; lanes<16 hold K {0..7,16..23},
// lanes>=16 hold K {8..15,24..31}; halves are K-sequential in each 8-half chunk.
__device__ __forceinline__ v16h load_a_frag(const _Float16* base, int stride, int k0, int lane) {
    const int row = lane & 15;
    const int ko  = (lane >> 4) * 8;
    const _Float16* p = base + row * stride + k0 + ko;
    v8h lo = *(const v8h*)p;          // K = k0+ko .. +7
    v8h hi = *(const v8h*)(p + 16);   // K = k0+16+ko .. +7
    v16h r;
#pragma unroll
    for (int i = 0; i < 8; ++i) { r[i] = lo[i]; r[8 + i] = hi[i]; }
    return r;
}

// B fragment (32x16, f16). Per ISA: n = lane&15; lanes<16 hold K {0..7,16..23},
// lanes>=16 hold K {8..15,24..31}; halves K-sequential. Row-major [n][k] source.
__device__ __forceinline__ v16h load_b_frag(const _Float16* base, int stride, int n0, int k0, int lane) {
    const int col = lane & 15;
    const int ko  = (lane >> 4) * 8;
    const _Float16* p = base + (size_t)(n0 + col) * stride + k0 + ko;
    v8h lo = *(const v8h*)p;
    v8h hi = *(const v8h*)(p + 16);
    v16h r;
#pragma unroll
    for (int i = 0; i < 8; ++i) { r[i] = lo[i]; r[8 + i] = hi[i]; }
    return r;
}

#define WMMA_F16(a, b, c) \
    __builtin_amdgcn_wmma_f32_16x16x32_f16(false, (a), false, (b), (short)0, (c), false, false)

// ---------------- setup kernels ----------------

__global__ void cvt_wih_kernel(const float* __restrict__ W, _Float16* __restrict__ Wh) {
    int idx = blockIdx.x * 256 + threadIdx.x;        // GDIM*HDIM threads, cols 0..511 only
    int g = idx >> 9;
    int d = idx & 511;
    Wh[(size_t)g * WIH_STRIDE + d] = (_Float16)W[(size_t)g * DDIM + d];
}

__global__ void cvt_fcw_kernel(const float* __restrict__ W, _Float16* __restrict__ Wh) {
    int idx = blockIdx.x * 256 + threadIdx.x;        // FCW_ROWS*HDIM threads
    int n = idx >> 9;
    int k = idx & 511;
    _Float16 v = (_Float16)0.0f;
    if (n < DDIM) v = (_Float16)W[(size_t)n * HDIM + k];
    Wh[(size_t)n * HDIM + k] = v;
}

// hb[b][g] = sum_k h0[b,k] * W_hh[g,k] + b_ih[g] + b_hh[g]
__global__ void hb_kernel(const float* __restrict__ h0, const float* __restrict__ W_hh,
                          const float* __restrict__ b_ih, const float* __restrict__ b_hh,
                          float* __restrict__ hb) {
    __shared__ float hs[HDIM];
    const int b = blockIdx.x;                         // 64
    const int g = blockIdx.y * 256 + threadIdx.x;     // gridDim.y = 8
    for (int i = threadIdx.x; i < HDIM; i += 256) hs[i] = h0[b * HDIM + i];
    __syncthreads();
    float acc = b_ih[g] + b_hh[g];
    const float* w = W_hh + (size_t)g * HDIM;
#pragma unroll 4
    for (int k = 0; k < HDIM; ++k) acc += hs[k] * w[k];
    hb[(size_t)b * GDIM + g] = acc;
}

// ---------------- fused main kernel ----------------

__global__ __launch_bounds__(128) void decoder_main_kernel(
    const float*    __restrict__ X,        // [65536, 513] (B*T flattened)
    const float*    __restrict__ W_ih_f32, // [2048, 513] (only col 512 read here)
    const _Float16* __restrict__ wih_h,    // [2048, 520] f16
    const float*    __restrict__ c0,       // [64, 512]
    const float*    __restrict__ hbp,      // [64, 2048]
    const _Float16* __restrict__ fcw_h,    // [528, 512] f16 (rows >=513 zero)
    const float*    __restrict__ fc_b,     // [513]
    float*          __restrict__ out)      // [65536, 513]
{
    __shared__ __align__(16) _Float16 Xh[RT * XH_STRIDE];
    __shared__ __align__(16) _Float16 Hh[RT * HH_STRIDE];
    __shared__ float xlast[RT];

    const int tid  = threadIdx.x;
    const int lane = tid & 31;
    const int wave = tid >> 5;
    const int row0 = blockIdx.x * RT;
    const int bidx = row0 >> 10;            // all RT rows share one batch index
    const int lcol = lane & 15;             // N index of C/D fragment
    const int mhi  = (lane >> 4) * 8;       // M offset of C/D fragment

    // ---- stage X tile (f32 -> f16 into LDS), coalesced ----
    for (int e = tid; e < RT * DDIM; e += 128) {
        int r = e / DDIM;
        int d = e - r * DDIM;
        float v = X[(size_t)row0 * DDIM + e];
        if (d < HDIM) Xh[r * XH_STRIDE + d] = (_Float16)v;
        else          xlast[r] = v;         // column 512 kept in f32
    }
    __syncthreads();

    const v8f vzero = {0.f, 0.f, 0.f, 0.f, 0.f, 0.f, 0.f, 0.f};

    // ---- phase 1: gates GEMM + LSTM cell -> h_new (f16) in LDS ----
    for (int ht = wave; ht < 32; ht += 4) {
        v8f acc[2][4];
#pragma unroll
        for (int rt = 0; rt < 2; ++rt)
#pragma unroll
            for (int gg = 0; gg < 4; ++gg) acc[rt][gg] = vzero;

        for (int k0 = 0; k0 < HDIM; k0 += 32) {
            v16h a0 = load_a_frag(Xh,                  XH_STRIDE, k0, lane);
            v16h a1 = load_a_frag(Xh + 16 * XH_STRIDE, XH_STRIDE, k0, lane);
#pragma unroll
            for (int gg = 0; gg < 4; ++gg) {
                v16h b = load_b_frag(wih_h, WIH_STRIDE, gg * HDIM + ht * 16, k0, lane);
                acc[0][gg] = WMMA_F16(a0, b, acc[0][gg]);
                acc[1][gg] = WMMA_F16(a1, b, acc[1][gg]);
            }
        }

        // rank-1 update for input column d=512, plus hidden bias, per gate column
        float w512[4], bh[4];
#pragma unroll
        for (int gg = 0; gg < 4; ++gg) {
            int ncol = gg * HDIM + ht * 16 + lcol;
            w512[gg] = W_ih_f32[(size_t)ncol * DDIM + 512];
            bh[gg]   = hbp[(size_t)bidx * GDIM + ncol];
        }
        float cv = c0[bidx * HDIM + ht * 16 + lcol];

#pragma unroll
        for (int rt = 0; rt < 2; ++rt) {
#pragma unroll
            for (int v = 0; v < 8; ++v) {
                int m = rt * 16 + v + mhi;
                float xl = xlast[m];
                float gi = sigmoid_(acc[rt][0][v] + xl * w512[0] + bh[0]);
                float gf = sigmoid_(acc[rt][1][v] + xl * w512[1] + bh[1]);
                float gG = tanh_   (acc[rt][2][v] + xl * w512[2] + bh[2]);
                float go = sigmoid_(acc[rt][3][v] + xl * w512[3] + bh[3]);
                float cn = gf * cv + gi * gG;
                float hn = go * tanh_(cn);
                Hh[m * HH_STRIDE + ht * 16 + lcol] = (_Float16)hn;
            }
        }
    }
    __syncthreads();

    // ---- phase 2: out = h_new @ fc_w^T + fc_b ----
    for (int nt = wave; nt < NT_OUT; nt += 4) {
        v8f o0 = vzero, o1 = vzero;
        for (int k0 = 0; k0 < HDIM; k0 += 32) {
            v16h b  = load_b_frag(fcw_h, HDIM, nt * 16, k0, lane);
            v16h a0 = load_a_frag(Hh,                  HH_STRIDE, k0, lane);
            v16h a1 = load_a_frag(Hh + 16 * HH_STRIDE, HH_STRIDE, k0, lane);
            o0 = WMMA_F16(a0, b, o0);
            o1 = WMMA_F16(a1, b, o1);
        }
        int n = nt * 16 + lcol;
        if (n < DDIM) {
            float bb = fc_b[n];
#pragma unroll
            for (int v = 0; v < 8; ++v) {
                out[(size_t)(row0 +      v + mhi) * DDIM + n] = o0[v] + bb;
                out[(size_t)(row0 + 16 + v + mhi) * DDIM + n] = o1[v] + bb;
            }
        }
    }
}

// ---------------- launch ----------------

extern "C" void kernel_launch(void* const* d_in, const int* in_sizes, int n_in,
                              void* d_out, int out_size, void* d_ws, size_t ws_size,
                              hipStream_t stream) {
    (void)in_sizes; (void)n_in; (void)out_size; (void)ws_size;
    const float* X    = (const float*)d_in[0];
    const float* h0   = (const float*)d_in[1];
    const float* c0   = (const float*)d_in[2];
    const float* W_ih = (const float*)d_in[3];
    const float* W_hh = (const float*)d_in[4];
    const float* b_ih = (const float*)d_in[5];
    const float* b_hh = (const float*)d_in[6];
    const float* fc_w = (const float*)d_in[7];
    const float* fc_b = (const float*)d_in[8];
    float* out = (float*)d_out;

    char* ws = (char*)d_ws;
    _Float16* wih_h = (_Float16*)ws;                                // 2048*520*2 = 2,129,920 B
    size_t off = (size_t)GDIM * WIH_STRIDE * sizeof(_Float16);
    _Float16* fcw_h = (_Float16*)(ws + off);                        // 528*512*2 = 540,672 B
    off += (size_t)FCW_ROWS * HDIM * sizeof(_Float16);
    float* hb = (float*)(ws + off);                                 // 64*2048*4 = 524,288 B

    cvt_wih_kernel<<<(GDIM * HDIM) / 256, 256, 0, stream>>>(W_ih, wih_h);
    cvt_fcw_kernel<<<(FCW_ROWS * HDIM) / 256, 256, 0, stream>>>(fc_w, fcw_h);
    hb_kernel<<<dim3(64, GDIM / 256), 256, 0, stream>>>(h0, W_hh, b_ih, b_hh, hb);

    const int M = 64 * 1024;                 // B*T rows
    decoder_main_kernel<<<M / RT, 128, 0, stream>>>(
        X, W_ih, wih_h, c0, hb, fcw_h, fc_b, out);
}
